// GMN_54099408060560
// MI455X (gfx1250) — matrix operand order repaired and checked by reading, not compiled, not run on hardware
//
#include <hip/hip_runtime.h>
#include <hip/hip_bf16.h>

// MeshGraphNet-style GNN for MI455X (gfx1250), wave32 + WMMA f16.
// All GEMMs are [rows,K] x [K,128] done with v_wmma_f32_16x16x32_f16.
// One wave computes a 16-row x 128-col output stripe (8 C tiles of 16x16).
// Weights converted once per launch to f16, transposed to [N][K] so B
// fragments are contiguous loads. Gather (x[senders]/x[receivers]) fused
// into edge-MLP layer 0; segment_sum fused into edge-MLP layer 3 store via
// global_atomic_add_f32; LayerNorm fused via half-wave __shfl_xor reductions.
// Workspace requirement: ~361 MB.

typedef __attribute__((ext_vector_type(16))) _Float16 v16h;
typedef __attribute__((ext_vector_type(8)))  _Float16 v8h;
typedef __attribute__((ext_vector_type(8)))  float    v8f;

#define N_NODES 50000
#define N_EDGES 500000
#define RN_PAD  50048    // roundup(N_NODES, 128)
#define RE_PAD  500096   // roundup(N_EDGES, 128)
#define N_MP    15

union HF16 { v16h v; struct { v8h lo, hi; } p; };

// A-matrix 16x32 f16 fragment (ISA 7.12.2): lanes 0-15 hold K {kb..kb+7, kb+16..kb+23},
// lanes 16-31 hold K {kb+8..kb+15, kb+24..kb+31}. rowp already includes row*stride.
__device__ __forceinline__ v16h frag_a(const _Float16* rowp, int kb, int half) {
  const _Float16* p = rowp + kb + half * 8;
  HF16 f;
  f.p.lo = *(const v8h*)(p);
  f.p.hi = *(const v8h*)(p + 16);
  return f.v;
}

// B-matrix 32x16 f16 fragment: lane = column, lanes 0-15 hold K {kb..kb+7, kb+16..kb+23},
// lanes 16-31 hold K {kb+8..kb+15, kb+24..kb+31} (same k order as frag_a).
// colp already includes column*K.
__device__ __forceinline__ v16h frag_b(const _Float16* colp, int kb, int half) {
  const _Float16* p = colp + kb + half * 8;
  HF16 f;
  f.p.lo = *(const v8h*)(p);
  f.p.hi = *(const v8h*)(p + 16);
  return f.v;
}

// Y[R,128] = f(A[R,K] @ Wt^T + bias), optional ReLU / LayerNorm / scatter-add.
// GATHER: 0 = A rows from X (row stride = K)
//         1 = edge concat: kt<4 -> X[idxS[row]], kt<8 -> X[idxR[row]], else X2[row] (stride 128)
//         2 = node concat: kt<4 -> X[row], else X2[row] (stride 128)
// SCATTER: also atomicAdd f32 row into agg[idxR[row]] (segment_sum fusion).
template<int KTILES, int GATHER, bool RELU, bool LN, bool SCATTER>
__global__ __launch_bounds__(256, 1)
void k_mlp(const _Float16* __restrict__ X, const _Float16* __restrict__ X2,
           const int* __restrict__ idxS, const int* __restrict__ idxR,
           const _Float16* __restrict__ Wt, const float* __restrict__ bias,
           const float* __restrict__ gamma, const float* __restrict__ beta,
           _Float16* __restrict__ Y, float* __restrict__ agg, int R)
{
  constexpr int K = KTILES * 32;
  const int lane = threadIdx.x & 31;
  const int wave = threadIdx.x >> 5;
  const int r    = lane & 15;
  const int half = lane >> 4;
  const int m0   = (blockIdx.x * 8 + wave) * 16;
  const int arow = m0 + r;
  const bool aok = (arow < R);

  float bcol[8];
#pragma unroll
  for (int t = 0; t < 8; ++t) bcol[t] = bias[t * 16 + r];

  v8f acc[8];
#pragma unroll
  for (int t = 0; t < 8; ++t) acc[t] = v8f{0.f,0.f,0.f,0.f,0.f,0.f,0.f,0.f};

  const _Float16* colp = Wt + (size_t)r * K;

#pragma unroll
  for (int kt = 0; kt < KTILES; ++kt) {
    const _Float16* ab;
    int kb;
    if constexpr (GATHER == 0) {
      ab = X + (size_t)arow * K; kb = kt * 32;
    } else if constexpr (GATHER == 1) {
      if (kt < 4)      { ab = X + (size_t)(aok ? idxS[arow] : 0) * 128; kb = kt * 32; }
      else if (kt < 8) { ab = X + (size_t)(aok ? idxR[arow] : 0) * 128; kb = (kt - 4) * 32; }
      else             { ab = X2 + (size_t)arow * 128;                  kb = (kt - 8) * 32; }
    } else {
      if (kt < 4) { ab = X  + (size_t)arow * 128; kb = kt * 32; }
      else        { ab = X2 + (size_t)arow * 128; kb = (kt - 4) * 32; }
    }
    const v16h a = frag_a(ab, kb, half);
#pragma unroll
    for (int t = 0; t < 8; ++t) {
      const v16h b = frag_b(colp + (size_t)t * 16 * K, kt * 32, half);
      acc[t] = __builtin_amdgcn_wmma_f32_16x16x32_f16(
          false, a, false, b, (short)0, acc[t], false, false);
    }
  }

#pragma unroll
  for (int t = 0; t < 8; ++t) {
#pragma unroll
    for (int v = 0; v < 8; ++v) {
      float x = acc[t][v] + bcol[t];
      if constexpr (RELU) x = fmaxf(x, 0.f);
      acc[t][v] = x;
    }
  }

  if constexpr (LN) {
    float gcol[8], ecol[8];
#pragma unroll
    for (int t = 0; t < 8; ++t) { gcol[t] = gamma[t * 16 + r]; ecol[t] = beta[t * 16 + r]; }
    // C/D layout: row (v + 8*half) lives in vgpr v across lanes of one 16-lane half;
    // xor masks 1..8 reduce within each half.
#pragma unroll
    for (int v = 0; v < 8; ++v) {
      float s = 0.f, q = 0.f;
#pragma unroll
      for (int t = 0; t < 8; ++t) { float x = acc[t][v]; s += x; q += x * x; }
#pragma unroll
      for (int m = 1; m < 16; m <<= 1) {
        s += __shfl_xor(s, m, 32);
        q += __shfl_xor(q, m, 32);
      }
      const float mean = s * (1.f / 128.f);
      const float var  = q * (1.f / 128.f) - mean * mean;
      const float rstd = rsqrtf(var + 1e-5f);
#pragma unroll
      for (int t = 0; t < 8; ++t)
        acc[t][v] = (acc[t][v] - mean) * rstd * gcol[t] + ecol[t];
    }
  }

#pragma unroll
  for (int v = 0; v < 8; ++v) {
    const int row = m0 + v + half * 8;
    if (row < R) {
      int rcv = 0;
      if constexpr (SCATTER) rcv = idxR[row];
#pragma unroll
      for (int t = 0; t < 8; ++t) {
        const int col = t * 16 + r;
        const float x = acc[t][v];
        Y[(size_t)row * 128 + col] = (_Float16)x;
        if constexpr (SCATTER) atomicAdd(agg + (size_t)rcv * 128 + col, x);
      }
    }
  }
}

// Decoder final layer: [R,128] x [128,12] (N padded to 16); emits cols 1..2 as f32.
__global__ __launch_bounds__(256, 1)
void k_dec_final(const _Float16* __restrict__ X, const _Float16* __restrict__ Wt,
                 const float* __restrict__ bias, float* __restrict__ out, int R)
{
  const int lane = threadIdx.x & 31;
  const int wave = threadIdx.x >> 5;
  const int r    = lane & 15;
  const int half = lane >> 4;
  const int m0   = (blockIdx.x * 8 + wave) * 16;
  const int arow = m0 + r;

  v8f acc = v8f{0.f,0.f,0.f,0.f,0.f,0.f,0.f,0.f};
  const _Float16* colp = Wt + (size_t)r * 128;
#pragma unroll
  for (int kt = 0; kt < 4; ++kt) {
    const v16h a = frag_a(X + (size_t)arow * 128, kt * 32, half);
    const v16h b = frag_b(colp, kt * 32, half);
    acc = __builtin_amdgcn_wmma_f32_16x16x32_f16(false, a, false, b, (short)0, acc, false, false);
  }
  if (r == 1 || r == 2) {
    const float bc = bias[r];
#pragma unroll
    for (int v = 0; v < 8; ++v) {
      const int row = m0 + v + half * 8;
      if (row < R) out[(size_t)row * 2 + (r - 1)] = acc[v] + bc;
    }
  }
}

// W f32 [K,N] row-major -> Wt f16 [NP][KP] (transposed, zero-padded). Grid = NP*KP/256.
__global__ void k_wconv(const float* __restrict__ W, _Float16* __restrict__ Wt,
                        int K, int N, int KP)
{
  const int i = blockIdx.x * blockDim.x + threadIdx.x;
  const int n = i / KP;
  const int k = i - n * KP;
  const float v = (k < K && n < N) ? W[(size_t)k * N + n] : 0.f;
  Wt[i] = (_Float16)v;
}

// f32 [R,DIN] -> f16 [Rpad,32] zero-padded. Grid = Rpad*32/256.
__global__ void k_inconv(const float* __restrict__ src, _Float16* __restrict__ dst,
                         int R, int DIN)
{
  const int i = blockIdx.x * blockDim.x + threadIdx.x;
  const int row = i >> 5, c = i & 31;
  const float v = (row < R && c < DIN) ? src[(size_t)row * DIN + c] : 0.f;
  dst[i] = (_Float16)v;
}

__global__ void k_f2h(const float* __restrict__ s, _Float16* __restrict__ d, size_t n)
{
  const size_t i = (size_t)blockIdx.x * blockDim.x + threadIdx.x;
  if (i < n) d[i] = (_Float16)s[i];
}

extern "C" void kernel_launch(void* const* d_in, const int* in_sizes, int n_in,
                              void* d_out, int out_size, void* d_ws, size_t ws_size,
                              hipStream_t stream)
{
  (void)in_sizes; (void)n_in; (void)out_size; (void)ws_size;

  const float* node_attr = (const float*)d_in[0];
  const float* edge_attr = (const float*)d_in[1];
  const int*   senders   = (const int*)d_in[2];
  const int*   receivers = senders + N_EDGES;
  // Param layout (setup_inputs dict order, MLP = Ws[4], bs[4], ln?):
  //  3..12  node_enc (W0..3, b0..3, g, beta)
  // 13..22  edge_enc
  // 23..30  dec (W0..3, b0..3; no ln)
  // 31+20i  mp[i].edge_upd ; +10 mp[i].node_upd

  char* p = (char*)d_ws;
  auto take = [&p](size_t bytes) -> void* {
    void* q = (void*)p; p += (bytes + 255) & ~(size_t)255; return q;
  };
  _Float16* xin   = (_Float16*)take((size_t)RN_PAD * 32  * 2);
  _Float16* ein   = (_Float16*)take((size_t)RE_PAD * 32  * 2);
  _Float16* X0    = (_Float16*)take((size_t)RN_PAD * 128 * 2);
  _Float16* X1    = (_Float16*)take((size_t)RN_PAD * 128 * 2);
  _Float16* E0    = (_Float16*)take((size_t)RE_PAD * 128 * 2);
  _Float16* E1    = (_Float16*)take((size_t)RE_PAD * 128 * 2);
  float*    agg   = (float*)   take((size_t)RN_PAD * 128 * 4);
  _Float16* agg16 = (_Float16*)take((size_t)RN_PAD * 128 * 2);

  auto conv_w = [&](int pidx, _Float16* dst, int K, int N, int KP, int NP) {
    k_wconv<<<(NP * KP) / 256, 256, 0, stream>>>((const float*)d_in[pidx], dst, K, N, KP);
  };
  auto B = [&](int pidx) { return (const float*)d_in[pidx]; };

  _Float16 *Wne[4], *Wee[4], *Wdec[4], *Wme[N_MP][4], *Wmn[N_MP][4];
  {
    const int Ks[4] = {12, 128, 128, 128};
    for (int j = 0; j < 4; ++j) {
      const int KP = (j == 0) ? 32 : 128;
      Wne[j] = (_Float16*)take((size_t)128 * KP * 2);
      conv_w(3 + j, Wne[j], Ks[j], 128, KP, 128);
    }
  }
  {
    const int Ks[4] = {3, 128, 128, 128};
    for (int j = 0; j < 4; ++j) {
      const int KP = (j == 0) ? 32 : 128;
      Wee[j] = (_Float16*)take((size_t)128 * KP * 2);
      conv_w(13 + j, Wee[j], Ks[j], 128, KP, 128);
    }
  }
  for (int j = 0; j < 4; ++j) {
    const int N  = (j == 3) ? 12 : 128;
    const int NP = (j == 3) ? 16 : 128;
    Wdec[j] = (_Float16*)take((size_t)NP * 128 * 2);
    conv_w(23 + j, Wdec[j], 128, N, 128, NP);
  }
  for (int i = 0; i < N_MP; ++i) {
    const int eb = 31 + 20 * i, nb = eb + 10;
    const int Ke[4] = {384, 128, 128, 128};
    const int Kn[4] = {256, 128, 128, 128};
    for (int j = 0; j < 4; ++j) {
      Wme[i][j] = (_Float16*)take((size_t)128 * Ke[j] * 2);
      conv_w(eb + j, Wme[i][j], Ke[j], 128, Ke[j], 128);
      Wmn[i][j] = (_Float16*)take((size_t)128 * Kn[j] * 2);
      conv_w(nb + j, Wmn[i][j], Kn[j], 128, Kn[j], 128);
    }
  }

  const int GN = RN_PAD / 128;
  const int GE = RE_PAD / 128;

  k_inconv<<<(RN_PAD * 32) / 256, 256, 0, stream>>>(node_attr, xin, N_NODES, 12);
  k_inconv<<<(RE_PAD * 32) / 256, 256, 0, stream>>>(edge_attr, ein, N_EDGES, 3);

  // node encoder: xin -> X1 -> X0 -> X1 -> X0 (LN)
  k_mlp<1,0,true ,false,false><<<GN,256,0,stream>>>(xin, nullptr, nullptr, nullptr, Wne[0], B(7),  nullptr, nullptr, X1, nullptr, N_NODES);
  k_mlp<4,0,true ,false,false><<<GN,256,0,stream>>>(X1,  nullptr, nullptr, nullptr, Wne[1], B(8),  nullptr, nullptr, X0, nullptr, N_NODES);
  k_mlp<4,0,true ,false,false><<<GN,256,0,stream>>>(X0,  nullptr, nullptr, nullptr, Wne[2], B(9),  nullptr, nullptr, X1, nullptr, N_NODES);
  k_mlp<4,0,false,true ,false><<<GN,256,0,stream>>>(X1,  nullptr, nullptr, nullptr, Wne[3], B(10), B(11),   B(12),   X0, nullptr, N_NODES);

  // edge encoder: ein -> E1 -> E0 -> E1 -> E0 (LN)
  k_mlp<1,0,true ,false,false><<<GE,256,0,stream>>>(ein, nullptr, nullptr, nullptr, Wee[0], B(17), nullptr, nullptr, E1, nullptr, N_EDGES);
  k_mlp<4,0,true ,false,false><<<GE,256,0,stream>>>(E1,  nullptr, nullptr, nullptr, Wee[1], B(18), nullptr, nullptr, E0, nullptr, N_EDGES);
  k_mlp<4,0,true ,false,false><<<GE,256,0,stream>>>(E0,  nullptr, nullptr, nullptr, Wee[2], B(19), nullptr, nullptr, E1, nullptr, N_EDGES);
  k_mlp<4,0,false,true ,false><<<GE,256,0,stream>>>(E1,  nullptr, nullptr, nullptr, Wee[3], B(20), B(21),   B(22),   E0, nullptr, N_EDGES);

  for (int i = 0; i < N_MP; ++i) {
    const int eb = 31 + 20 * i, nb = eb + 10;
    hipMemsetAsync(agg, 0, (size_t)RN_PAD * 128 * 4, stream);
    // edge MLP: concat(x[s], x[r], e) -> E1 -> E0 -> E1 -> E0 (LN + scatter segment_sum)
    k_mlp<12,1,true ,false,false><<<GE,256,0,stream>>>(X0, E0, senders, receivers, Wme[i][0], B(eb+4), nullptr, nullptr, E1, nullptr, N_EDGES);
    k_mlp< 4,0,true ,false,false><<<GE,256,0,stream>>>(E1, nullptr, nullptr, nullptr,          Wme[i][1], B(eb+5), nullptr, nullptr, E0, nullptr, N_EDGES);
    k_mlp< 4,0,true ,false,false><<<GE,256,0,stream>>>(E0, nullptr, nullptr, nullptr,          Wme[i][2], B(eb+6), nullptr, nullptr, E1, nullptr, N_EDGES);
    k_mlp< 4,0,false,true ,true ><<<GE,256,0,stream>>>(E1, nullptr, nullptr, receivers,        Wme[i][3], B(eb+7), B(eb+8), B(eb+9), E0, agg,     N_EDGES);
    k_f2h<<<(RN_PAD * 128) / 256, 256, 0, stream>>>(agg, agg16, (size_t)RN_PAD * 128);
    // node MLP: concat(x, agg) -> X1 -> X0 -> X1 -> X0 (LN)
    k_mlp< 8,2,true ,false,false><<<GN,256,0,stream>>>(X0, agg16, nullptr, nullptr, Wmn[i][0], B(nb+4), nullptr, nullptr, X1, nullptr, N_NODES);
    k_mlp< 4,0,true ,false,false><<<GN,256,0,stream>>>(X1, nullptr, nullptr, nullptr, Wmn[i][1], B(nb+5), nullptr, nullptr, X0, nullptr, N_NODES);
    k_mlp< 4,0,true ,false,false><<<GN,256,0,stream>>>(X0, nullptr, nullptr, nullptr, Wmn[i][2], B(nb+6), nullptr, nullptr, X1, nullptr, N_NODES);
    k_mlp< 4,0,false,true ,false><<<GN,256,0,stream>>>(X1, nullptr, nullptr, nullptr, Wmn[i][3], B(nb+7), B(nb+8), B(nb+9), X0, nullptr, N_NODES);
  }

  // decoder: X0 -> X1 -> X0 -> X1 -> d_out[:,1:3]
  k_mlp<4,0,true,false,false><<<GN,256,0,stream>>>(X0, nullptr, nullptr, nullptr, Wdec[0], B(27), nullptr, nullptr, X1, nullptr, N_NODES);
  k_mlp<4,0,true,false,false><<<GN,256,0,stream>>>(X1, nullptr, nullptr, nullptr, Wdec[1], B(28), nullptr, nullptr, X0, nullptr, N_NODES);
  k_mlp<4,0,true,false,false><<<GN,256,0,stream>>>(X0, nullptr, nullptr, nullptr, Wdec[2], B(29), nullptr, nullptr, X1, nullptr, N_NODES);
  k_dec_final<<<GN,256,0,stream>>>(X1, Wdec[3], B(30), (float*)d_out, N_NODES);
}
